// MainModel_85383949844915
// MI455X (gfx1250) — matrix operand order repaired and checked by reading, not compiled, not run on hardware
//
#include <hip/hip_runtime.h>

typedef __attribute__((ext_vector_type(16))) __bf16 v16bf;
typedef __attribute__((ext_vector_type(8)))  float  v8f;
typedef int v4i __attribute__((vector_size(16)));

#define L_SEQ 768
#define DM 256
#define DFF 1024
#define NHEAD 8
#define NLAYER 8

#if defined(__HIP_DEVICE_COMPILE__) && __has_builtin(__builtin_amdgcn_global_load_async_to_lds_b128)
#define HAVE_ASYNC_LDS 1
#else
#define HAVE_ASYNC_LDS 0
#endif

union U16BF { v16bf v; uint4 q[2]; };

__device__ inline v8f wmma_bf16(v16bf a, v16bf b, v8f c) {
  return __builtin_amdgcn_wmma_f32_16x16x32_bf16(false, a, false, b, (short)0, c, false, false);
}

// ---- A operand: rows row0+lane%16, K chunk at base k0 (row-major src, leading dim ld)
__device__ inline v16bf loadA(const __bf16* __restrict__ src, int ld, int row, int k0, int hi) {
  U16BF u;
  const __bf16* p = src + (size_t)row * ld + k0;
  u.q[0] = *(const uint4*)(p + hi * 8);
  u.q[1] = *(const uint4*)(p + 16 + hi * 8);
  return u.v;
}
// ---- B operand from transposed weights WT[outCol][K]: col = n-index, K chunk k0
__device__ inline v16bf loadB(const __bf16* __restrict__ srcT, int ldk, int col, int k0, int hi) {
  U16BF u;
  const __bf16* p = srcT + (size_t)col * ldk + k0 + hi * 8;
  u.q[0] = *(const uint4*)(p);
  u.q[1] = *(const uint4*)(p + 16);
  return u.v;
}

// ===================== prep kernels =====================
__global__ void bucket_kernel(int* __restrict__ lut) {
  int d = blockIdx.x * blockDim.x + threadIdx.x;
  if (d >= 1535) return;
  int rel = d - 767;
  int ret = rel < 0 ? 32 : 0;
  int arp = rel < 0 ? -rel : rel;
  int bv;
  if (arp < 16) bv = arp;
  else {
    int vil = 16 + (int)(logf((float)arp * (1.0f / 16.0f)) * (16.0f / 2.7725887222397811f));
    bv = vil < 31 ? vil : 31;
  }
  lut[d] = ret + bv;
}

__global__ void convT_kernel(const float* __restrict__ W, __bf16* __restrict__ WT,
                             int R, int C, int total) {
  for (int idx = blockIdx.x * blockDim.x + threadIdx.x; idx < total;
       idx += gridDim.x * blockDim.x) {
    int rc = R * C;
    int l = idx / rc; int rem = idx - l * rc;
    int r = rem / C;  int c = rem - r * C;
    WT[(size_t)l * rc + (size_t)c * R + r] = (__bf16)W[idx];
  }
}

__global__ void embed_kernel(const int* __restrict__ seq, const float* __restrict__ emb,
                             float* __restrict__ x) {
  int i = blockIdx.x * blockDim.x + threadIdx.x;
  if (i >= L_SEQ * DM) return;
  int l = i >> 8, d = i & 255;
  x[i] = emb[seq[l] * DM + d];
}

// ===================== layernorm (f32 in, bf16 out) =====================
__global__ __launch_bounds__(256)
void ln_bf16_kernel(const float* __restrict__ x, const float* __restrict__ s,
                    const float* __restrict__ b, __bf16* __restrict__ out) {
  __shared__ float red[256];
  int row = blockIdx.x, t = threadIdx.x;
  float v = x[row * DM + t];
  red[t] = v; __syncthreads();
  #pragma unroll
  for (int off = 128; off > 0; off >>= 1) { if (t < off) red[t] += red[t + off]; __syncthreads(); }
  float mean = red[0] * (1.0f / DM);
  __syncthreads();
  float d = v - mean;
  red[t] = d * d; __syncthreads();
  #pragma unroll
  for (int off = 128; off > 0; off >>= 1) { if (t < off) red[t] += red[t + off]; __syncthreads(); }
  float var = red[0] * (1.0f / DM);
  out[row * DM + t] = (__bf16)(d * rsqrtf(var + 1e-5f) * s[t] + b[t]);
}

// ===================== generic WMMA GEMM =====================
// out[M,N] = A[M,K](bf16) @ BT[N,K](bf16)^T ; optional bias, gelu, f32 residual-add, bf16 store
__global__ __launch_bounds__(128)
void gemm_bf16_kernel(const __bf16* __restrict__ A, const __bf16* __restrict__ BT,
                      const float* __restrict__ bias, float* __restrict__ resid,
                      __bf16* __restrict__ outb, int M, int N, int K,
                      int act, int storeT, int nWaveTiles) {
  int wid = blockIdx.x * 4 + (threadIdx.x >> 5);
  if (wid >= nWaveTiles) return;
  int Nt = N >> 6;
  int mt = wid / Nt, nt = wid - mt * Nt;
  int lane = threadIdx.x & 31, lo = lane & 15, hi = lane >> 4;
  int m0 = mt * 16, n0 = nt * 64;
  v8f acc0 = {}, acc1 = {}, acc2 = {}, acc3 = {};
  for (int kc = 0; kc < K; kc += 32) {
    v16bf a = loadA(A, K, m0 + lo, kc, hi);
    acc0 = wmma_bf16(a, loadB(BT, K, n0 + lo,      kc, hi), acc0);
    acc1 = wmma_bf16(a, loadB(BT, K, n0 + 16 + lo, kc, hi), acc1);
    acc2 = wmma_bf16(a, loadB(BT, K, n0 + 32 + lo, kc, hi), acc2);
    acc3 = wmma_bf16(a, loadB(BT, K, n0 + 48 + lo, kc, hi), acc3);
  }
  v8f accs[4] = {acc0, acc1, acc2, acc3};
  #pragma unroll
  for (int c = 0; c < 4; ++c) {
    int col = n0 + c * 16 + lo;
    float bv = bias ? bias[col] : 0.0f;
    #pragma unroll
    for (int r = 0; r < 8; ++r) {
      int row = m0 + r + hi * 8;
      float v = accs[c][r] + bv;
      if (act == 1) v = 0.5f * v * (1.0f + erff(v * 0.70710678118f));
      if (resid) {
        float nv = resid[(size_t)row * DM + col] + v;
        resid[(size_t)row * DM + col] = nv;
        v = nv;
      }
      if (outb) {
        if (storeT) outb[(size_t)col * M + row] = (__bf16)v;
        else        outb[(size_t)row * N + col] = (__bf16)v;
      }
    }
  }
}

// ===================== flash attention (1 wave = 1 head x 16 rows) =====================
__global__ __launch_bounds__(32)
void attn_kernel(const __bf16* __restrict__ q, const __bf16* __restrict__ k,
                 const __bf16* __restrict__ vT, const float* __restrict__ rp_emb,
                 const int* __restrict__ blut, __bf16* __restrict__ ao) {
  __shared__ float bias_lut[1535];
  __shared__ __bf16 p_lds[16 * 32];
  int h = blockIdx.x & 7;
  int it = blockIdx.x >> 3;
  int lane = threadIdx.x, lo = lane & 15, hi = lane >> 4;
  for (int d = lane; d < 1535; d += 32) bias_lut[d] = rp_emb[blut[d] * NHEAD + h];
  __syncthreads();
  int i0 = it * 16;
  const float rscale = 0.17677669529663687f; // 1/sqrt(32)
  v16bf qa = loadA(q, DM, i0 + lo, h * 32, hi);
  float m[8], s[8];
  v8f o0 = {}, o1 = {};
  #pragma unroll
  for (int r = 0; r < 8; ++r) { m[r] = -1e30f; s[r] = 0.0f; }
  for (int j0 = 0; j0 < L_SEQ; j0 += 32) {
    v16bf kb0 = loadB(k, DM, j0 + lo,      h * 32, hi);
    v16bf kb1 = loadB(k, DM, j0 + 16 + lo, h * 32, hi);
    v8f z = {};
    v8f sc0 = wmma_bf16(qa, kb0, z);
    v8f sc1 = wmma_bf16(qa, kb1, z);
    #pragma unroll
    for (int r = 0; r < 8; ++r) {
      int i = i0 + r + hi * 8;
      float s0 = sc0[r] * rscale + bias_lut[i - (j0 + lo) + 767];
      float s1 = sc1[r] * rscale + bias_lut[i - (j0 + 16 + lo) + 767];
      float lm = fmaxf(s0, s1);
      lm = fmaxf(lm, __shfl_xor(lm, 1, 16));
      lm = fmaxf(lm, __shfl_xor(lm, 2, 16));
      lm = fmaxf(lm, __shfl_xor(lm, 4, 16));
      lm = fmaxf(lm, __shfl_xor(lm, 8, 16));
      float nm = fmaxf(m[r], lm);
      float resc = __expf(m[r] - nm);
      float p0 = __expf(s0 - nm);
      float p1 = __expf(s1 - nm);
      float ls = p0 + p1;
      ls += __shfl_xor(ls, 1, 16);
      ls += __shfl_xor(ls, 2, 16);
      ls += __shfl_xor(ls, 4, 16);
      ls += __shfl_xor(ls, 8, 16);
      s[r] = s[r] * resc + ls;
      o0[r] *= resc; o1[r] *= resc;
      m[r] = nm;
      int prow = r + hi * 8;
      p_lds[prow * 32 + lo]      = (__bf16)p0;
      p_lds[prow * 32 + 16 + lo] = (__bf16)p1;
    }
    __syncthreads();
    U16BF up;
    up.q[0] = *(const uint4*)(p_lds + lo * 32 + hi * 8);
    up.q[1] = *(const uint4*)(p_lds + lo * 32 + 16 + hi * 8);
    __syncthreads();
    v16bf vb0 = loadB(vT, L_SEQ, h * 32 + lo,      j0, hi);
    v16bf vb1 = loadB(vT, L_SEQ, h * 32 + 16 + lo, j0, hi);
    o0 = wmma_bf16(up.v, vb0, o0);
    o1 = wmma_bf16(up.v, vb1, o1);
  }
  #pragma unroll
  for (int r = 0; r < 8; ++r) {
    int row = i0 + r + hi * 8;
    float inv = 1.0f / s[r];
    ao[(size_t)row * DM + h * 32 + lo]      = (__bf16)(o0[r] * inv);
    ao[(size_t)row * DM + h * 32 + 16 + lo] = (__bf16)(o1[r] * inv);
  }
}

// ===================== fused pair head =====================
// 1 wave = 32 pairs (fixed i, 32 consecutive j) = two 16-pair A-register sets, so
// every cls_w1 B-tile read from LDS feeds TWO wmmas (halves LDS read traffic/FLOP).
// cls_w1^T staged through 32KB LDS in 4 x 64-column groups shared by 8 waves/WG,
// using GLOBAL_LOAD_ASYNC_TO_LDS when the toolchain exposes it.
__global__ __launch_bounds__(256)
void pair_kernel(const __bf16* __restrict__ pq, const __bf16* __restrict__ pk,
                 const float* __restrict__ prp, const int* __restrict__ blut,
                 const __bf16* __restrict__ w1T, const float* __restrict__ b1,
                 const float* __restrict__ w2, const float* __restrict__ b2,
                 float* __restrict__ out) {
  __shared__ __bf16 lds_w1[64 * 256];
  int wave = threadIdx.x >> 5, lane = threadIdx.x & 31, lo = lane & 15, hi = lane >> 4;
  int tid = blockIdx.x * 8 + wave;        // 18432 jobs: i in [0,768), jt in [0,24)
  int i = tid / 24, jt = tid - i * 24;
  int j0 = jt * 32;
  int jA = j0 + lo, jB = j0 + 16 + lo;
  const __bf16* pqrow = pq + (size_t)i * DM;
  const __bf16* pkA = pk + (size_t)jA * DM;
  const __bf16* pkB = pk + (size_t)jB * DM;
  const float*  prA = prp + (size_t)blut[i - jA + 767] * DM;
  const float*  prB = prp + (size_t)blut[i - jB + 767] * DM;
  v16bf A0[8], A1[8];
  #pragma unroll
  for (int kc = 0; kc < 8; ++kc) {
    int base = kc * 32;
    U16BF uq;
    uq.q[0] = *(const uint4*)(pqrow + base + hi * 8);
    uq.q[1] = *(const uint4*)(pqrow + base + 16 + hi * 8);
    {
      U16BF uk;
      uk.q[0] = *(const uint4*)(pkA + base + hi * 8);
      uk.q[1] = *(const uint4*)(pkA + base + 16 + hi * 8);
      const float* pr0 = prA + base + hi * 8;
      const float* pr1 = prA + base + 16 + hi * 8;
      v16bf a;
      #pragma unroll
      for (int e = 0; e < 8; ++e)  a[e] = (__bf16)((float)uq.v[e] * (float)uk.v[e] + pr0[e]);
      #pragma unroll
      for (int e = 8; e < 16; ++e) a[e] = (__bf16)((float)uq.v[e] * (float)uk.v[e] + pr1[e - 8]);
      A0[kc] = a;
    }
    {
      U16BF uk;
      uk.q[0] = *(const uint4*)(pkB + base + hi * 8);
      uk.q[1] = *(const uint4*)(pkB + base + 16 + hi * 8);
      const float* pr0 = prB + base + hi * 8;
      const float* pr1 = prB + base + 16 + hi * 8;
      v16bf a;
      #pragma unroll
      for (int e = 0; e < 8; ++e)  a[e] = (__bf16)((float)uq.v[e] * (float)uk.v[e] + pr0[e]);
      #pragma unroll
      for (int e = 8; e < 16; ++e) a[e] = (__bf16)((float)uq.v[e] * (float)uk.v[e] + pr1[e - 8]);
      A1[kc] = a;
    }
  }
  float a00[8] = {}, a01[8] = {}, a10[8] = {}, a11[8] = {};
  for (int g = 0; g < 4; ++g) {
    __syncthreads();
    const uint4* src = (const uint4*)(w1T + (size_t)g * 64 * 256);
    uint4* dst = (uint4*)lds_w1;
#if HAVE_ASYNC_LDS
    for (int t = threadIdx.x; t < 4096; t += 256)
      __builtin_amdgcn_global_load_async_to_lds_b128(
          (__attribute__((address_space(1))) v4i*)(v4i*)(src + t),
          (__attribute__((address_space(3))) v4i*)(v4i*)(dst + t), 0, 0);
#if __has_builtin(__builtin_amdgcn_s_wait_asynccnt)
    __builtin_amdgcn_s_wait_asynccnt(0);
#else
    asm volatile("s_wait_asynccnt 0" ::: "memory");
#endif
#else
    for (int t = threadIdx.x; t < 4096; t += 256) dst[t] = src[t];
#endif
    if (g < 3) __builtin_prefetch(w1T + (size_t)(g + 1) * 64 * 256, 0, 1);
    __syncthreads();
    #pragma unroll
    for (int nt = 0; nt < 4; ++nt) {
      v8f c0 = {}, c1 = {};
      #pragma unroll
      for (int kc = 0; kc < 8; ++kc) {
        U16BF ub;
        const __bf16* p = lds_w1 + (size_t)(nt * 16 + lo) * 256 + kc * 32 + hi * 8;
        ub.q[0] = *(const uint4*)(p);
        ub.q[1] = *(const uint4*)(p + 16);
        c0 = wmma_bf16(A0[kc], ub.v, c0);
        c1 = wmma_bf16(A1[kc], ub.v, c1);
      }
      int col = g * 64 + nt * 16 + lo;
      float bb = b1[col], w20 = w2[col * 2], w21 = w2[col * 2 + 1];
      #pragma unroll
      for (int r = 0; r < 8; ++r) {
        float h0 = c0[r] + bb; h0 = h0 > 0.0f ? h0 : 0.0f;
        float h1 = c1[r] + bb; h1 = h1 > 0.0f ? h1 : 0.0f;
        a00[r] += h0 * w20; a01[r] += h0 * w21;
        a10[r] += h1 * w20; a11[r] += h1 * w21;
      }
    }
  }
  #pragma unroll
  for (int r = 0; r < 8; ++r) {
    #pragma unroll
    for (int off = 1; off < 16; off <<= 1) {
      a00[r] += __shfl_xor(a00[r], off, 16);
      a01[r] += __shfl_xor(a01[r], off, 16);
      a10[r] += __shfl_xor(a10[r], off, 16);
      a11[r] += __shfl_xor(a11[r], off, 16);
    }
  }
  if (lo == 0) {
    #pragma unroll
    for (int r = 0; r < 8; ++r) {
      int p = r + hi * 8;
      size_t oA = ((size_t)i * L_SEQ + j0 + p) * 2;
      size_t oB = ((size_t)i * L_SEQ + j0 + 16 + p) * 2;
      out[oA]     = a00[r] + b2[0];
      out[oA + 1] = a01[r] + b2[1];
      out[oB]     = a10[r] + b2[0];
      out[oB + 1] = a11[r] + b2[1];
    }
  }
}

// ===================== host launcher =====================
extern "C" void kernel_launch(void* const* d_in, const int* in_sizes, int n_in,
                              void* d_out, int out_size, void* d_ws, size_t ws_size,
                              hipStream_t stream) {
  const int*   seq     = (const int*)  d_in[0];
  const float* tok_emb = (const float*)d_in[1];
  const float* rp_emb  = (const float*)d_in[2];
  const float* wq      = (const float*)d_in[3];
  const float* wk      = (const float*)d_in[4];
  const float* wv      = (const float*)d_in[5];
  const float* wo      = (const float*)d_in[6];
  const float* ln1_s   = (const float*)d_in[7];
  const float* ln1_b   = (const float*)d_in[8];
  const float* ln2_s   = (const float*)d_in[9];
  const float* ln2_b   = (const float*)d_in[10];
  const float* fw1     = (const float*)d_in[11];
  const float* fb1     = (const float*)d_in[12];
  const float* fw2     = (const float*)d_in[13];
  const float* fb2     = (const float*)d_in[14];
  const float* lnf_s   = (const float*)d_in[15];
  const float* lnf_b   = (const float*)d_in[16];
  const float* pqw     = (const float*)d_in[17];
  const float* pqb     = (const float*)d_in[18];
  const float* pkw     = (const float*)d_in[19];
  const float* pkb     = (const float*)d_in[20];
  const float* prp     = (const float*)d_in[21];
  const float* c1w     = (const float*)d_in[22];
  const float* c1b     = (const float*)d_in[23];
  const float* c2w     = (const float*)d_in[24];
  const float* c2b     = (const float*)d_in[25];
  float* out = (float*)d_out;

  char* w = (char*)d_ws; size_t off = 0;
  auto alloc = [&](size_t bytes) -> void* {
    void* p = w + off; off = (off + bytes + 255) & ~(size_t)255; return p;
  };
  float*  x    = (float*) alloc((size_t)L_SEQ * DM * 4);
  __bf16* h    = (__bf16*)alloc((size_t)L_SEQ * DM * 2);
  __bf16* qb   = (__bf16*)alloc((size_t)L_SEQ * DM * 2);
  __bf16* kb   = (__bf16*)alloc((size_t)L_SEQ * DM * 2);
  __bf16* vT   = (__bf16*)alloc((size_t)DM * L_SEQ * 2);
  __bf16* ao   = (__bf16*)alloc((size_t)L_SEQ * DM * 2);
  __bf16* fbuf = (__bf16*)alloc((size_t)L_SEQ * DFF * 2);
  __bf16* pq16 = (__bf16*)alloc((size_t)L_SEQ * DM * 2);
  __bf16* pk16 = (__bf16*)alloc((size_t)L_SEQ * DM * 2);
  int*    lut  = (int*)   alloc(1535 * 4);
  const size_t WSZ = (size_t)DM * DM;      // 65536
  const size_t FSZ = (size_t)DM * DFF;     // 262144
  __bf16* wqT  = (__bf16*)alloc(NLAYER * WSZ * 2);
  __bf16* wkT  = (__bf16*)alloc(NLAYER * WSZ * 2);
  __bf16* wvT  = (__bf16*)alloc(NLAYER * WSZ * 2);
  __bf16* woT  = (__bf16*)alloc(NLAYER * WSZ * 2);
  __bf16* fw1T = (__bf16*)alloc(NLAYER * FSZ * 2);
  __bf16* fw2T = (__bf16*)alloc(NLAYER * FSZ * 2);
  __bf16* pqwT = (__bf16*)alloc(WSZ * 2);
  __bf16* pkwT = (__bf16*)alloc(WSZ * 2);
  __bf16* c1wT = (__bf16*)alloc(WSZ * 2);

  // ---- prep: bucket LUT + weight convert/transpose to bf16 [out][in]
  bucket_kernel<<<6, 256, 0, stream>>>(lut);
  convT_kernel<<<2048, 256, 0, stream>>>(wq,  wqT,  DM, DM, (int)(NLAYER * WSZ));
  convT_kernel<<<2048, 256, 0, stream>>>(wk,  wkT,  DM, DM, (int)(NLAYER * WSZ));
  convT_kernel<<<2048, 256, 0, stream>>>(wv,  wvT,  DM, DM, (int)(NLAYER * WSZ));
  convT_kernel<<<2048, 256, 0, stream>>>(wo,  woT,  DM, DM, (int)(NLAYER * WSZ));
  convT_kernel<<<4096, 256, 0, stream>>>(fw1, fw1T, DM, DFF, (int)(NLAYER * FSZ));
  convT_kernel<<<4096, 256, 0, stream>>>(fw2, fw2T, DFF, DM, (int)(NLAYER * FSZ));
  convT_kernel<<<256,  256, 0, stream>>>(pqw, pqwT, DM, DM, (int)WSZ);
  convT_kernel<<<256,  256, 0, stream>>>(pkw, pkwT, DM, DM, (int)WSZ);
  convT_kernel<<<256,  256, 0, stream>>>(c1w, c1wT, DM, DM, (int)WSZ);

  embed_kernel<<<L_SEQ, 256, 0, stream>>>(seq, tok_emb, x);

  for (int l = 0; l < NLAYER; ++l) {
    ln_bf16_kernel<<<L_SEQ, 256, 0, stream>>>(x, ln1_s + l * DM, ln1_b + l * DM, h);
    // q,k: [768,256]; v stored transposed [256,768] for the P@V B-operand
    gemm_bf16_kernel<<<48, 128, 0, stream>>>(h, wqT + l * WSZ, nullptr, nullptr, qb,
                                             L_SEQ, DM, DM, 0, 0, 192);
    gemm_bf16_kernel<<<48, 128, 0, stream>>>(h, wkT + l * WSZ, nullptr, nullptr, kb,
                                             L_SEQ, DM, DM, 0, 0, 192);
    gemm_bf16_kernel<<<48, 128, 0, stream>>>(h, wvT + l * WSZ, nullptr, nullptr, vT,
                                             L_SEQ, DM, DM, 0, 1, 192);
    attn_kernel<<<NHEAD * 48, 32, 0, stream>>>(qb, kb, vT, rp_emb, lut, ao);
    // x += ao @ Wo
    gemm_bf16_kernel<<<48, 128, 0, stream>>>(ao, woT + l * WSZ, nullptr, x, nullptr,
                                             L_SEQ, DM, DM, 0, 0, 192);
    ln_bf16_kernel<<<L_SEQ, 256, 0, stream>>>(x, ln2_s + l * DM, ln2_b + l * DM, h);
    // ffn1 + exact gelu
    gemm_bf16_kernel<<<192, 128, 0, stream>>>(h, fw1T + l * FSZ, fb1 + l * DFF, nullptr,
                                              fbuf, L_SEQ, DFF, DM, 1, 0, 768);
    // x += ffn2
    gemm_bf16_kernel<<<48, 128, 0, stream>>>(fbuf, fw2T + l * FSZ, fb2 + l * DM, x, nullptr,
                                             L_SEQ, DM, DFF, 0, 0, 192);
  }

  ln_bf16_kernel<<<L_SEQ, 256, 0, stream>>>(x, lnf_s, lnf_b, h);
  gemm_bf16_kernel<<<48, 128, 0, stream>>>(h, pqwT, pqb, nullptr, pq16,
                                           L_SEQ, DM, DM, 0, 0, 192);
  gemm_bf16_kernel<<<48, 128, 0, stream>>>(h, pkwT, pkb, nullptr, pk16,
                                           L_SEQ, DM, DM, 0, 0, 192);
  // fused pair head: 768*24 = 18432 32-pair jobs, 8 waves/WG
  pair_kernel<<<2304, 256, 0, stream>>>(pq16, pk16, prp, lut, c1wT, c1b, c2w, c2b, out);
}
